// Head_73452530696597
// MI455X (gfx1250) — compile-verified, matrix operand order checked by reading, not run on hardware
//
#include <hip/hip_runtime.h>

// ---------------------------------------------------------------------------
// Sliding-window attention head for MI455X (gfx1250), wave32 + bf16 WMMA.
// v3: M=32 projection tiles (2x weight reuse), branchless hot loops with
//     uniform-branch boundary fixup, all operands moved as 128-bit lines.
//   prep:  WT = W^T (bf16), biasT = bias^T (bf16), zero vT
//   k = x @ Wk (row-major bf16) ; vT = (x @ Wv)^T (d-major bf16, padded S)
//   q = x[-512:] @ Wq ; qb[b,w,t] = q . bias[:,t]
//   attn: dense 16x528 QK band per 16-row q tile, shifted softmax,
//         dense 16x544 AV GEMM against vT (zero-padded band).
// ---------------------------------------------------------------------------

typedef __attribute__((ext_vector_type(16))) __bf16 v16bf;
typedef __attribute__((ext_vector_type(8)))  __bf16 v8bf;
typedef __attribute__((ext_vector_type(8)))  float  v8f;
typedef __attribute__((ext_vector_type(4)))  float  v4f;

#define NEMBD 1024
#define HS    64
#define T     512
#define SROWS 1023   // valid x rows per batch
#define SPAD  1024   // padded k rows per batch (row 1023 zeroed)
#define VPAD  1056   // padded vT columns per (b,d): 1024 + 32 slack, zeroed
#define NB    4
#define WPAD  544    // 528-wide score band padded to a multiple of 32

static __device__ __forceinline__ __bf16 bf(float f) { return (__bf16)f; }

// convert 8 contiguous fp32 (two 16B loads) -> v8bf
static __device__ __forceinline__ v8bf cvt8(const float* __restrict__ p) {
  const v4f a = *(const v4f*)p;
  const v4f b = *(const v4f*)(p + 4);
  v8bf r;
  r[0] = (__bf16)a[0]; r[1] = (__bf16)a[1]; r[2] = (__bf16)a[2]; r[3] = (__bf16)a[3];
  r[4] = (__bf16)b[0]; r[5] = (__bf16)b[1]; r[6] = (__bf16)b[2]; r[7] = (__bf16)b[3];
  return r;
}

static __device__ __forceinline__ v16bf cat16(v8bf lo, v8bf hi) {
  return __builtin_shufflevector(lo, hi, 0, 1, 2, 3, 4, 5, 6, 7, 8, 9, 10, 11,
                                 12, 13, 14, 15);
}

static __device__ __forceinline__ v16bf ld16(const __bf16* __restrict__ p) {
  return cat16(*(const v8bf*)p, *(const v8bf*)(p + 16));
}

static __device__ __forceinline__ v8f wmma_bf16(v16bf a, v16bf b, v8f c) {
  // (neg_a, A, neg_b, B, c_mod, C, reuse_a, reuse_b)
  return __builtin_amdgcn_wmma_f32_16x16x32_bf16(false, a, false, b, (short)0,
                                                 c, false, false);
}

// ---------------- Stage 0: transpose/convert shared operands ---------------
// WT: [3][64][1024] bf16 (wsel,d,c) ; biasT: [512][64] bf16 ; vT zero-fill.
__global__ void prep_kernel(const float* __restrict__ Wk,
                            const float* __restrict__ Wv,
                            const float* __restrict__ Wq,
                            const float* __restrict__ bias,
                            __bf16* __restrict__ WT,
                            __bf16* __restrict__ biasT,
                            __bf16* __restrict__ vT) {
  const int NW    = 3 * HS * NEMBD;  // 196608
  const int NBIAS = T * HS;          // 32768
  const int NVZ   = NB * HS * VPAD;  // 270336
  const int idx   = blockIdx.x * blockDim.x + threadIdx.x;
  if (idx < NW) {
    const int wsel = idx / (HS * NEMBD);
    const int rem  = idx % (HS * NEMBD);
    const int d    = rem / NEMBD;
    const int c    = rem % NEMBD;
    const float* W = (wsel == 0) ? Wk : ((wsel == 1) ? Wv : Wq);
    WT[idx] = bf(W[(size_t)c * HS + d]);
  } else if (idx < NW + NBIAS) {
    const int r = idx - NW;
    const int t = r / HS;
    const int d = r % HS;
    biasT[r] = bf(bias[(size_t)d * T + t]);
  } else if (idx < NW + NBIAS + NVZ) {
    vT[idx - NW - NBIAS] = bf(0.0f);
  }
}

// ---------------- Stage 1: k (row-major) and vT (d-major) projections -------
// M=32 per wave: two 16-row A tiles share each weight B tile.
__global__ __launch_bounds__(32) void proj_kv_kernel(
    const float* __restrict__ x, const __bf16* __restrict__ WT,
    __bf16* __restrict__ kbf, __bf16* __restrict__ vT) {
  const int lane  = threadIdx.x;
  const int m     = lane & 15;        // A row / B,C column within tile
  const int hi8   = (lane >> 4) * 8;  // K sub-offset for upper half-wave
  const int mbase = (lane >> 4) * 8;  // C row base
  const int tile  = blockIdx.x;       // 0..31 (rows tile*32 .. +31)
  const int b     = blockIdx.y;
  const int isV   = blockIdx.z;       // 0 -> k, 1 -> v
  const __bf16* Wt = WT + (size_t)isV * HS * NEMBD;  // [64][1024]

  const int rowA = tile * 32 + m;
  const int rowB = rowA + 16;  // only tile 31, m 15 exceeds SROWS-1
  const float* xrowA = x + ((size_t)b * SROWS + rowA) * NEMBD;
  const float* xrowB =
      x + ((size_t)b * SROWS + (rowB < SROWS ? rowB : 0)) * NEMBD;

  v8f acc[2][4] = {};  // [half][nt]; garbage row 1023 fixed up after stores
  for (int kc = 0; kc < NEMBD; kc += 32) {
    const v16bf aA = cat16(cvt8(xrowA + kc + hi8), cvt8(xrowA + kc + 16 + hi8));
    const v16bf aB = cat16(cvt8(xrowB + kc + hi8), cvt8(xrowB + kc + 16 + hi8));
#pragma unroll
    for (int nt = 0; nt < 4; ++nt) {
      const v16bf bm = ld16(Wt + (size_t)(nt * 16 + m) * NEMBD + kc + hi8);
      acc[0][nt] = wmma_bf16(aA, bm, acc[0][nt]);
      acc[1][nt] = wmma_bf16(aB, bm, acc[1][nt]);
    }
  }
  if (isV == 0) {
#pragma unroll
    for (int half = 0; half < 2; ++half)
#pragma unroll
      for (int nt = 0; nt < 4; ++nt) {
        const int col = nt * 16 + m;
#pragma unroll
        for (int g = 0; g < 8; ++g) {
          const int orow = tile * 32 + half * 16 + mbase + g;
          kbf[((size_t)b * SPAD + orow) * HS + col] = bf(acc[half][nt][g]);
        }
      }
    if (tile == 31 && lane < 8) {  // uniform-ish fixup: zero k row 1023
      const v8bf z = {};
      *(v8bf*)&kbf[((size_t)b * SPAD + (SPAD - 1)) * HS + lane * 8] = z;
    }
  } else {
#pragma unroll
    for (int half = 0; half < 2; ++half)
#pragma unroll
      for (int nt = 0; nt < 4; ++nt) {
        const int col = nt * 16 + m;  // d index
        v8bf r;
#pragma unroll
        for (int g = 0; g < 8; ++g) r[g] = bf(acc[half][nt][g]);
        // transposed store: contiguous 8 bf16 along s
        *(v8bf*)&vT[((size_t)b * HS + col) * VPAD + tile * 32 + half * 16 +
                    mbase] = r;
      }
    if (tile == 31) {  // zero vT column 1023 (s = 1023) for all d
#pragma unroll
      for (int d = lane; d < HS; d += 32)
        vT[((size_t)b * HS + d) * VPAD + (SPAD - 1)] = bf(0.0f);
    }
  }
}

// ---------------- Stage 2: q projection (x rows 511..1022), M=32 ------------
__global__ __launch_bounds__(32) void proj_q_kernel(
    const float* __restrict__ x, const __bf16* __restrict__ WqT,
    __bf16* __restrict__ qbf) {
  const int lane  = threadIdx.x;
  const int m     = lane & 15;
  const int hi8   = (lane >> 4) * 8;
  const int mbase = (lane >> 4) * 8;
  const int tile  = blockIdx.x;  // 0..15 (q rows tile*32 .. +31)
  const int b     = blockIdx.y;

  const int xrA = 511 + tile * 32 + m;  // <= 1006
  const float* xrowA = x + ((size_t)b * SROWS + xrA) * NEMBD;
  const float* xrowB = xrowA + (size_t)16 * NEMBD;  // <= 1022, always valid

  v8f acc[2][4] = {};
  for (int kc = 0; kc < NEMBD; kc += 32) {
    const v16bf aA = cat16(cvt8(xrowA + kc + hi8), cvt8(xrowA + kc + 16 + hi8));
    const v16bf aB = cat16(cvt8(xrowB + kc + hi8), cvt8(xrowB + kc + 16 + hi8));
#pragma unroll
    for (int nt = 0; nt < 4; ++nt) {
      const v16bf bm = ld16(WqT + (size_t)(nt * 16 + m) * NEMBD + kc + hi8);
      acc[0][nt] = wmma_bf16(aA, bm, acc[0][nt]);
      acc[1][nt] = wmma_bf16(aB, bm, acc[1][nt]);
    }
  }
#pragma unroll
  for (int half = 0; half < 2; ++half)
#pragma unroll
    for (int nt = 0; nt < 4; ++nt)
#pragma unroll
      for (int g = 0; g < 8; ++g)
        qbf[((size_t)b * T + tile * 32 + half * 16 + mbase + g) * HS +
            nt * 16 + m] = bf(acc[half][nt][g]);
}

// ---------------- Stage 3: qb[b,w,t] = q . bias[:,t]  (K=64) ----------------
__global__ __launch_bounds__(32) void qb_kernel(
    const __bf16* __restrict__ qbf, const __bf16* __restrict__ biasT,
    float* __restrict__ qb) {
  const int lane  = threadIdx.x;
  const int m     = lane & 15;
  const int hi8   = (lane >> 4) * 8;
  const int mbase = (lane >> 4) * 8;
  const int wt    = blockIdx.x;  // 0..31
  const int b     = blockIdx.y;
  const int w0    = wt * 16;

  const __bf16* qrow = qbf + ((size_t)b * T + w0 + m) * HS;
  const v16bf a0 = ld16(qrow + hi8);
  const v16bf a1 = ld16(qrow + 32 + hi8);

  for (int nt = 0; nt < 32; ++nt) {  // 32 t-tiles of 16
    const int t0 = nt * 16;
    const __bf16* br = biasT + (size_t)(t0 + m) * HS;  // [t][d]
    v8f acc = {};
    acc = wmma_bf16(a0, ld16(br + hi8), acc);
    acc = wmma_bf16(a1, ld16(br + 32 + hi8), acc);
#pragma unroll
    for (int g = 0; g < 8; ++g)
      qb[((size_t)b * T + w0 + mbase + g) * T + t0 + m] = acc[g];
  }
}

// ---------------- Stage 4: banded attention per (b, 16-row w tile) ----------
__global__ __launch_bounds__(32) void attn_kernel(
    const __bf16* __restrict__ qbf, const __bf16* __restrict__ kbf,
    const __bf16* __restrict__ vT, const float* __restrict__ qb,
    float* __restrict__ out) {
  __shared__ float  sQK[16][WPAD];  // 34816 B : QK band, cols s-w0 in [0,528)
  __shared__ __bf16 sP[16][WPAD];   // 17408 B : shifted probabilities

  const int lane  = threadIdx.x;
  const int m     = lane & 15;
  const int hi8   = (lane >> 4) * 8;
  const int mbase = (lane >> 4) * 8;
  const int wt    = blockIdx.x;  // 0..31
  const int b     = blockIdx.y;
  const int w0    = wt * 16;

  {  // zero P band (shift padding + WPAD tail), 16B per store
    v8bf z = {};
    v8bf* p = (v8bf*)&sP[0][0];
    for (int i = lane; i < 16 * WPAD / 8; i += 32) p[i] = z;
  }

  const __bf16* qrow = qbf + ((size_t)b * T + w0 + m) * HS;
  const v16bf aq0 = ld16(qrow + hi8);
  const v16bf aq1 = ld16(qrow + 32 + hi8);

  // QK band: C[w][s-w0] = q . k[s], s in [w0, w0+528)
  for (int nt = 0; nt < 33; ++nt) {
    const int s0 = w0 + nt * 16;
    const __bf16* krow = kbf + ((size_t)b * SPAD + s0 + m) * HS;  // s0+m<=1023
    v8f acc = {};
    acc = wmma_bf16(aq0, ld16(krow + hi8), acc);
    acc = wmma_bf16(aq1, ld16(krow + 32 + hi8), acc);
    const int col = nt * 16 + m;
#pragma unroll
    for (int g = 0; g < 8; ++g) sQK[mbase + g][col] = acc[g];
  }
  __syncthreads();

  // Shifted softmax: row r reads cols r+t, adds qb, writes bf16 P at r+t.
  const float* qbbase = qb + (size_t)(b * T + w0) * T;
  for (int r = 0; r < 16; ++r) {
    const float* qbr = qbbase + (size_t)r * T;
    float vals[16];
    float mx = -3.0e38f;
#pragma unroll
    for (int i = 0; i < 16; ++i) {
      const int t = lane + i * 32;
      const float sc = sQK[r][r + t] + qbr[t];
      vals[i] = sc;
      mx = fmaxf(mx, sc);
    }
#pragma unroll
    for (int off = 16; off; off >>= 1) mx = fmaxf(mx, __shfl_xor(mx, off, 32));
    float sum = 0.0f;
#pragma unroll
    for (int i = 0; i < 16; ++i) {
      const float e = __expf(vals[i] - mx);
      vals[i] = e;
      sum += e;
    }
#pragma unroll
    for (int off = 16; off; off >>= 1) sum += __shfl_xor(sum, off, 32);
    const float inv = 1.0f / sum;
#pragma unroll
    for (int i = 0; i < 16; ++i) {
      const int t = lane + i * 32;
      sP[r][r + t] = bf(vals[i] * inv);
    }
  }
  __syncthreads();

  // AV: out[w][d] = P_band @ v[w0 .. w0+543] via vT (contiguous K loads)
  v8f oacc[4] = {};
  for (int kc = 0; kc < WPAD; kc += 32) {  // 17 K steps
    const v16bf ap = ld16(&sP[m][kc + hi8]);
#pragma unroll
    for (int nt = 0; nt < 4; ++nt) {
      const __bf16* vr =
          vT + ((size_t)b * HS + nt * 16 + m) * VPAD + w0 + kc + hi8;
      oacc[nt] = wmma_bf16(ap, ld16(vr), oacc[nt]);
    }
  }
#pragma unroll
  for (int nt = 0; nt < 4; ++nt)
#pragma unroll
    for (int g = 0; g < 8; ++g)
      out[((size_t)b * T + w0 + mbase + g) * HS + nt * 16 + m] = oacc[nt][g];
}

// ---------------------------------------------------------------------------
extern "C" void kernel_launch(void* const* d_in, const int* in_sizes, int n_in,
                              void* d_out, int out_size, void* d_ws,
                              size_t ws_size, hipStream_t stream) {
  const float* x    = (const float*)d_in[0];
  const float* Wk   = (const float*)d_in[1];
  const float* Wv   = (const float*)d_in[2];
  const float* Wq   = (const float*)d_in[3];
  const float* bias = (const float*)d_in[4];

  char* ws = (char*)d_ws;
  __bf16* kbf   = (__bf16*)(ws);                 // 4*1024*64*2   = 524288 B
  __bf16* vT    = (__bf16*)(ws + 524288);        // 4*64*1056*2   = 540672 B
  __bf16* qbf   = (__bf16*)(ws + 1064960);       // 4*512*64*2    = 262144 B
  float*  qbp   = (float*)(ws + 1327104);        // 4*512*512*4   = 4194304 B
  __bf16* WT    = (__bf16*)(ws + 5521408);       // 3*64*1024*2   = 393216 B
  __bf16* biasT = (__bf16*)(ws + 5914624);       // 512*64*2      = 65536 B
  float*  out   = (float*)d_out;

  const int prepN = 3 * HS * NEMBD + T * HS + NB * HS * VPAD;
  prep_kernel<<<(prepN + 255) / 256, 256, 0, stream>>>(Wk, Wv, Wq, bias, WT,
                                                       biasT, vT);
  proj_kv_kernel<<<dim3(32, NB, 2), 32, 0, stream>>>(x, WT, kbf, vT);
  proj_q_kernel<<<dim3(16, NB), 32, 0, stream>>>(x, WT + 2 * HS * NEMBD, qbf);
  qb_kernel<<<dim3(32, NB), 32, 0, stream>>>(qbf, biasT, qbp);
  attn_kernel<<<dim3(32, NB), 32, 0, stream>>>(qbf, kbf, vT, qbp, out);
}
